// DiceLossLayer_24163486008133
// MI455X (gfx1250) — compile-verified, matrix operand order checked by reading, not run on hardware
//
#include <hip/hip_runtime.h>
#include <math.h>

typedef __attribute__((ext_vector_type(16))) _Float16 v16h;
typedef __attribute__((ext_vector_type(8)))  float    v8f;

#define GRIDN  256
#define NPTS   128
#define NWAVES 8
#define SLICES 4               // blocks per batch
#define ROWS_PER_WAVE (GRIDN / (NWAVES * SLICES))   // 8 scanlines per wave

// Exact wave32 sum via WMMA: every lane replicates its value into all 16 f16
// halves of its A-operand slice; B = ones. Row m of D = 16*(v_m + v_{m+16})
// regardless of B layout. Lane n holds D[0..7][n], lane n+16 holds D[8..15][n],
// so sum-of-8-dwords + shfl_xor(16) gives 16 * (full wave sum) on every lane.
__device__ __forceinline__ float wave_sum_wmma(float x) {
  v16h a, bo;
#pragma unroll
  for (int i = 0; i < 16; ++i) { a[i] = (_Float16)x; bo[i] = (_Float16)1.0f; }
  v8f c = {};
  c = __builtin_amdgcn_wmma_f32_16x16x32_f16(false, a, false, bo,
                                             (short)0, c, false, false);
  float s = c[0] + c[1] + c[2] + c[3] + c[4] + c[5] + c[6] + c[7];
  s += __shfl_xor(s, 16, 32);
  return s * 0.0625f;   // /16
}

__global__ void dice_zero_ws(unsigned int* ws) {
  if (threadIdx.x < 3 * 64) ws[threadIdx.x] = 0u;
}

__global__ __launch_bounds__(256)
void dice_raster_kernel(const float* __restrict__ points,
                        const float* __restrict__ dmap,
                        unsigned int* __restrict__ ws) {
  __shared__ float2 pts_s[NPTS];
  __shared__ unsigned int hist[NWAVES][GRIDN];   // wave-private histograms
  __shared__ float wsum[NWAVES][3];

  const int b     = blockIdx.x >> 2;       // batch
  const int slice = blockIdx.x & 3;        // y-slice of 64 rows
  const int tid   = threadIdx.x;
  const int lane  = tid & 31;
  const int w     = tid >> 5;

  // Load + scale + clip the polygon once per block.
  if (tid < NPTS) {
    float px = points[((size_t)b * NPTS + tid) * 2 + 0] * 255.0f;
    float py = points[((size_t)b * NPTS + tid) * 2 + 1] * 255.0f;
    px = fminf(fmaxf(px, 0.0f), 255.0f);
    py = fminf(fmaxf(py, 0.0f), 255.0f);
    pts_s[tid] = make_float2(px, py);
  }
  __syncthreads();

  // 4 edges per lane held in registers: edge e pairs p[e] with p[e-1].
  float pix[4], piy[4], pjx[4], pjy[4];
#pragma unroll
  for (int k = 0; k < 4; ++k) {
    int e = lane + 32 * k;
    float2 pi = pts_s[e];
    float2 pj = pts_s[(e + NPTS - 1) & (NPTS - 1)];
    pix[k] = pi.x; piy[k] = pi.y; pjx[k] = pj.x; pjy[k] = pj.y;
  }

  unsigned int c_mask = 0, c_bin = 0, c_int = 0;
  const float* dbase = dmap + (size_t)b * GRIDN * GRIDN;

  for (int it = 0; it < ROWS_PER_WAVE; ++it) {
    const int y = slice * (GRIDN / SLICES) + w * ROWS_PER_WAVE + it;
    const float fy = (float)y;

    // Issue this row's pixel loads early (8 contiguous floats per lane).
    const float4* dp = (const float4*)(dbase + (size_t)y * GRIDN + lane * 8);
    float4 r0 = dp[0];
    float4 r1 = dp[1];
    if (it + 1 < ROWS_PER_WAVE)   // global_prefetch_b8 for next row
      __builtin_prefetch(dbase + (size_t)(y + 1) * GRIDN + lane * 8, 0, 1);

    // Clear wave-private histogram (in-order DS within a wave: no barrier).
#pragma unroll
    for (int j = 0; j < 8; ++j) hist[w][lane * 8 + j] = 0u;

    // Rasterize: histogram floor(x-intersection) of crossing edges.
#pragma unroll
    for (int k = 0; k < 4; ++k) {
      bool cross = (piy[k] < fy && pjy[k] >= fy) ||
                   (pjy[k] < fy && piy[k] >= fy);
      if (cross) {
        float xint = pix[k] + (fy - piy[k]) / (pjy[k] - piy[k]) * (pjx[k] - pix[k]);
        int bk = (int)floorf(xint);
        bk = min(max(bk, 0), GRIDN - 1);
        atomicAdd(&hist[w][bk], 1u);
      }
    }

    // f(t) = #crossings < t via lane-local + cross-lane exclusive scan.
    unsigned int h[8];
    int lsum = 0;
#pragma unroll
    for (int j = 0; j < 8; ++j) { h[j] = hist[w][lane * 8 + j]; lsum += (int)h[j]; }

    int inc = lsum;
#pragma unroll
    for (int d = 1; d < 32; d <<= 1) {
      int t = __shfl_up(inc, d, 32);
      if (lane >= d) inc += t;
    }
    const int base  = inc - lsum;
    const int count = __shfl(inc, 31, 32);
    const int hi2   = (count - 2) >> 1;   // last valid pair index

    const float dv[8] = {r0.x, r0.y, r0.z, r0.w, r1.x, r1.y, r1.z, r1.w};
    int f = base;                          // f(x) at x = lane*8
#pragma unroll
    for (int j = 0; j < 8; ++j) {
      const int fn = f + (int)h[j];        // f(x+1)
      const int lo = f >> 1;               // ceil((f-1)/2), f >= 0
      const int hi = min((fn - 1) >> 1, hi2);
      const int covered = (lo <= hi) ? 1 : 0;
      const int binv = (dv[j] * 255.0f <= 127.0f) ? 1 : 0;
      c_mask += covered;
      c_bin  += binv;
      c_int  += covered & binv;
      f = fn;
    }
  }

  // Wave reduction via v_wmma (integer-exact), then block reduction.
  const float sm = wave_sum_wmma((float)c_mask);
  const float sb = wave_sum_wmma((float)c_bin);
  const float si = wave_sum_wmma((float)c_int);

  if (lane == 0) { wsum[w][0] = sm; wsum[w][1] = sb; wsum[w][2] = si; }
  __syncthreads();
  if (tid == 0) {
    float St = 0.f, Sq = 0.f, In = 0.f;
    for (int i = 0; i < NWAVES; ++i) { St += wsum[i][0]; Sq += wsum[i][1]; In += wsum[i][2]; }
    atomicAdd(&ws[b * 3 + 0], (unsigned int)(St + 0.5f));
    atomicAdd(&ws[b * 3 + 1], (unsigned int)(Sq + 0.5f));
    atomicAdd(&ws[b * 3 + 2], (unsigned int)(In + 0.5f));
  }
}

__global__ void dice_finalize(const unsigned int* __restrict__ ws,
                              float* __restrict__ out) {
  __shared__ float part[64];
  const int t = threadIdx.x;
  if (t < 64) {
    const float St = (float)ws[3 * t + 0];
    const float Sq = (float)ws[3 * t + 1];
    const float In = (float)ws[3 * t + 2];
    const float dice = (2.0f * In + 1e-6f) / (St + Sq + 1e-6f);
    part[t] = 1.0f - dice;
  }
  __syncthreads();
  if (t == 0) {
    float s = 0.f;
    for (int i = 0; i < 64; ++i) s += part[i];
    out[0] = s * (1.0f / 64.0f);
  }
}

extern "C" void kernel_launch(void* const* d_in, const int* in_sizes, int n_in,
                              void* d_out, int out_size, void* d_ws, size_t ws_size,
                              hipStream_t stream) {
  const float* points = (const float*)d_in[0];   // (64,128,1,2) f32
  const float* dmap   = (const float*)d_in[1];   // (64,256,256,1) f32
  float* out          = (float*)d_out;           // scalar f32
  unsigned int* ws    = (unsigned int*)d_ws;     // 64*3 uint accumulators

  dice_zero_ws<<<1, 256, 0, stream>>>(ws);
  dice_raster_kernel<<<64 * SLICES, 256, 0, stream>>>(points, dmap, ws);
  dice_finalize<<<1, 64, 0, stream>>>(ws, out);
}
